// MoE_41394894799018
// MI455X (gfx1250) — matrix operand order repaired and checked by reading, not compiled, not run on hardware
//
#include <hip/hip_runtime.h>
#include <hip/hip_bf16.h>

typedef _Float16 v16h __attribute__((ext_vector_type(16)));
typedef float    v8f  __attribute__((ext_vector_type(8)));

static __device__ __forceinline__ v8f wmma_f16(v16h a, v16h b, v8f c) {
  // D = A(16x32 f16) * B(32x16 f16) + C(16x16 f32)   -> v_wmma_f32_16x16x32_f16
  return __builtin_amdgcn_wmma_f32_16x16x32_f16(false, a, false, b, (short)0, c, false, false);
}

// Async copy of one 16-byte chunk: global -> LDS, tracked by ASYNCcnt.
static __device__ __forceinline__ void async_copy_b128(unsigned ldsOff, const void* gptr) {
  unsigned long long ga = (unsigned long long)gptr;
  asm volatile("global_load_async_to_lds_b128 %0, %1, off"
               :: "v"(ldsOff), "v"(ga) : "memory");
}
static __device__ __forceinline__ void wait_asynccnt0() {
  asm volatile("s_wait_asynccnt 0x0" ::: "memory");
}

// ---------------- elementwise helpers ----------------
__global__ void k_f32_to_f16(const float* __restrict__ s, _Float16* __restrict__ d, long n) {
  long i = (long)blockIdx.x * blockDim.x + threadIdx.x;
  if (i < n) d[i] = (_Float16)s[i];
}

__global__ void k_zero_f16(_Float16* p, long n) {
  long i = (long)blockIdx.x * blockDim.x + threadIdx.x;
  if (i < n) p[i] = (_Float16)0.f;
}

__global__ void k_bn_fold(const float* __restrict__ cb, const float* __restrict__ g,
                          const float* __restrict__ bt, const float* __restrict__ m,
                          const float* __restrict__ v,
                          float* __restrict__ scale, float* __restrict__ bias, int co) {
  int i = blockIdx.x * blockDim.x + threadIdx.x;
  if (i < co) {
    float s = g[i] * rsqrtf(v[i] + 1e-5f);
    scale[i] = s;                                // applied to conv result
    bias[i]  = cb[i] * s + bt[i] - m[i] * s;     // (conv+b)*s + bt - m*s
  }
}

// ---------------- implicit-GEMM conv 3x3 SAME + BN + ReLU (WMMA) ----------------
// Templated on compile-time power-of-two dims so all tile decode is masks/shifts
// and the K loops fully unroll into straight-line v_wmma runs.
template <int Ci, int H, int W, int Co>
__global__ void __launch_bounds__(128)
k_conv_wmma(const _Float16* __restrict__ in, const _Float16* __restrict__ w,
            const float* __restrict__ scale, const float* __restrict__ bias,
            _Float16* __restrict__ out)
{
  constexpr int coTiles = Co >> 4;
  constexpr int xTiles  = W >> 4;
  constexpr long HW = (long)H * W;
  constexpr int kcN = (Ci + 31) >> 5;

  const int lane = threadIdx.x & 31;
  const int wid  = threadIdx.x >> 5;
  unsigned tile = blockIdx.x * 4u + wid;         // grid sized exactly

  unsigned ct = tile % coTiles; unsigned t2 = tile / coTiles;
  unsigned xt = t2 % xTiles;    t2 /= xTiles;
  unsigned y  = t2 % H;
  unsigned b  = t2 / H;

  const int lm = lane & 15;
  const int hi = lane >> 4;
  const int co = ct * 16 + lm;                   // A-fragment row (M)
  const int xp = xt * 16 + lm;                   // B-fragment col (N)
  const int ka0 = hi ? 16 : 0;                   // contiguous k order (matches B)
  const int ka1 = hi ? 24 : 8;
  const int kb0 = hi ? 16 : 0;                   // contiguous k order (matches A)

  v8f acc = {};
  for (int ky = 0; ky < 3; ++ky) {
    int yy = (int)y + ky - 1;
    bool yok = (yy >= 0) && (yy < H);
    for (int kx = 0; kx < 3; ++kx) {
      int xx = xp + kx - 1;
      bool xok = yok && (xx >= 0) && (xx < W);
      #pragma unroll
      for (int kc = 0; kc < kcN; ++kc) {
        int cib = kc << 5;
        const _Float16* wp = w + ((long)co * Ci + cib) * 9 + ky * 3 + kx;
        const _Float16* ip = in + ((long)b * Ci + cib) * HW + (long)yy * W + xx;
        v16h af, bf;
        #pragma unroll
        for (int i = 0; i < 8; ++i) {
          int k1 = ka0 + i, k2 = ka1 + i;
          af[i]     = (cib + k1 < Ci) ? wp[(long)k1 * 9] : (_Float16)0.f;
          af[i + 8] = (cib + k2 < Ci) ? wp[(long)k2 * 9] : (_Float16)0.f;
        }
        #pragma unroll
        for (int j = 0; j < 16; ++j) {
          int k = kb0 + j;
          bf[j] = (xok && (cib + k < Ci)) ? ip[(long)k * HW] : (_Float16)0.f;
        }
        acc = wmma_f16(af, bf, acc);
      }
    }
  }
  #pragma unroll
  for (int r = 0; r < 8; ++r) {
    int mo = ct * 16 + r + hi * 8;               // C/D layout: lanes16-31 -> M=8+r
    float vv = acc[r] * scale[mo] + bias[mo];
    vv = fmaxf(vv, 0.f);
    out[((long)b * Co + mo) * HW + (long)y * W + (xt * 16 + lm)] = (_Float16)vv;
  }
}

// ---------------- 2x2 max-pool (f16) ----------------
__global__ void k_maxpool2(const _Float16* __restrict__ in, _Float16* __restrict__ out,
                           int Bn, int Cc, int H, int W) {
  long n = (long)Bn * Cc * (H >> 1) * (W >> 1);
  long i = (long)blockIdx.x * blockDim.x + threadIdx.x;
  if (i >= n) return;
  int Wo = W >> 1, Ho = H >> 1;
  int xo = (int)(i % Wo); long t = i / Wo;
  int yo = (int)(t % Ho); t /= Ho;
  int c  = (int)(t % Cc); int b = (int)(t / Cc);
  const _Float16* p = in + (((long)b * Cc + c) * H + yo * 2) * W + xo * 2;
  float m = fmaxf(fmaxf((float)p[0], (float)p[1]), fmaxf((float)p[W], (float)p[W + 1]));
  out[i] = (_Float16)m;
}

// ---------------- fused 2x2 max-pool + global mean -> feats ----------------
__global__ void k_pool_avg(const _Float16* __restrict__ in, float* __restrict__ f32o,
                           _Float16* __restrict__ f16o, int Bn, int Cc, int H, int W) {
  int i = blockIdx.x * blockDim.x + threadIdx.x;
  if (i >= Bn * Cc) return;
  const _Float16* p = in + (long)i * H * W;
  float s = 0.f;
  for (int y = 0; y < H; y += 2)
    for (int x = 0; x < W; x += 2)
      s += fmaxf(fmaxf((float)p[y * W + x], (float)p[y * W + x + 1]),
                 fmaxf((float)p[(y + 1) * W + x], (float)p[(y + 1) * W + x + 1]));
  s *= (4.f / (float)(H * W));
  f32o[i] = s;
  f16o[i] = (_Float16)s;
}

// ---------------- scorer GEMV + softmax (E=8, tiny) ----------------
__global__ void k_scorer(const float* __restrict__ feats, const float* __restrict__ sw,
                         const float* __restrict__ sb, float* __restrict__ scores) {
  int b = blockIdx.x * blockDim.x + threadIdx.x;
  if (b >= 512) return;
  float lg[8];
  #pragma unroll
  for (int e = 0; e < 8; ++e) lg[e] = sb[e];
  const float* f = feats + b * 256;
  for (int d = 0; d < 256; ++d) {
    float fv = f[d];
    const float* wr = sw + d * 8;
    #pragma unroll
    for (int e = 0; e < 8; ++e) lg[e] += fv * wr[e];
  }
  float mx = lg[0];
  #pragma unroll
  for (int e = 1; e < 8; ++e) mx = fmaxf(mx, lg[e]);
  float ex[8], den = 0.f;
  #pragma unroll
  for (int e = 0; e < 8; ++e) { ex[e] = expf(lg[e] - mx); den += ex[e]; }
  float inv = 1.f / den;
  #pragma unroll
  for (int e = 0; e < 8; ++e) scores[b * 8 + e] = ex[e] * inv;
}

// ---------------- capacity-constrained greedy router (single WG) ----------------
__global__ void __launch_bounds__(1024)
k_route(const float* __restrict__ scores, float* __restrict__ Dout,
        int* __restrict__ idx, float* __restrict__ validv)
{
  __shared__ unsigned long long keys[4096];
  __shared__ float D[4096];
  __shared__ int   ec[8];
  __shared__ int   sc[512];
  __shared__ float pen[8];
  const int tid = threadIdx.x;
  for (int i = tid; i < 4096; i += 1024) D[i] = 0.f;
  for (int i = tid; i < 512;  i += 1024) sc[i] = 0;
  if (tid < 8) { ec[tid] = 0; pen[tid] = 0.f; }
  __syncthreads();

  int total = 0;                                  // only lane 0 uses/updates
  for (int cyc = 0; cyc < 4; ++cyc) {
    for (int i = tid; i < 4096; i += 1024) {
      float s = scores[i] * (1.f - pen[i & 7]);
      unsigned u = __float_as_uint(s);
      u = (u & 0x80000000u) ? ~u : (u | 0x80000000u);   // order-preserving map
      keys[i] = (((unsigned long long)(~u)) << 32) | (unsigned)i; // asc => desc score
    }
    __syncthreads();
    for (int k = 2; k <= 4096; k <<= 1) {
      for (int j = k >> 1; j > 0; j >>= 1) {
        for (int i = tid; i < 4096; i += 1024) {
          int ixj = i ^ j;
          if (ixj > i) {
            unsigned long long a = keys[i], b = keys[ixj];
            bool up = ((i & k) == 0);
            if (up ? (a > b) : (a < b)) { keys[i] = b; keys[ixj] = a; }
          }
        }
        __syncthreads();
      }
    }
    if (tid == 0) {
      int assigned = 0;
      for (int t = 0; t < 4096; ++t) {
        int fi = (int)(keys[t] & 0xFFFFFFFFull);
        int bb = fi >> 3, ee = fi & 7;
        if (D[fi] == 0.f && ec[ee] < 128 && sc[bb] < 2 &&
            assigned < 256 && total < 1024) {
          D[fi] = 1.f; ec[ee]++; sc[bb]++; total++; assigned++;
        }
      }
    }
    __syncthreads();
    if (tid < 8) pen[tid] = (float)ec[tid] * (1.f / 128.f);
    __syncthreads();
  }
  for (int i = tid; i < 4096; i += 1024) Dout[i] = D[i];
  if (tid < 8) {                                  // per-expert compaction (top-C gather)
    int e = tid, cnt = 0;
    for (int b = 0; b < 512; ++b)
      if (D[b * 8 + e] > 0.f) { idx[e * 128 + cnt] = b; validv[e * 128 + cnt] = 1.f; cnt++; }
    for (; cnt < 128; ++cnt)  { idx[e * 128 + cnt] = 0; validv[e * 128 + cnt] = 0.f; }
  }
}

// ---------------- expert MLP layer 1: h = relu(gather(feats) @ ew1 + eb1) ----------------
// One block owns an (expert, n-tile) column: B panel (K=256 x 16) is async-copied
// to LDS once and reused by all 8 M-tiles (4 waves x 2 tiles each).
__global__ void __launch_bounds__(128)
k_expert_l1(const _Float16* __restrict__ feats16, const int* __restrict__ idx,
            const _Float16* __restrict__ w1, const float* __restrict__ b1,
            _Float16* __restrict__ h)
{
  __shared__ _Float16 smem[256 * 16];             // 8 KiB weight panel
  const int tid  = threadIdx.x;
  const int lane = tid & 31;
  const int wid  = tid >> 5;
  const int nt = blockIdx.x & 63;
  const int e  = blockIdx.x >> 6;
  const int n0 = nt * 16;

  {   // async stage: 256 rows x 32B = 512 chunks of 16B
    const _Float16* src = w1 + (long)e * 256 * 1024 + n0;
    unsigned ldsBase = (unsigned)(unsigned long long)smem;
    #pragma unroll
    for (int c = 0; c < 4; ++c) {
      int ch = tid + c * 128;
      int row = ch >> 1, half = ch & 1;
      async_copy_b128(ldsBase + row * 32 + half * 16, src + (long)row * 1024 + half * 8);
    }
    wait_asynccnt0();
  }
  __syncthreads();

  const int lm = lane & 15, hi = lane >> 4;
  const int ka0 = hi ? 16 : 0, ka1 = hi ? 24 : 8, kb0 = hi ? 16 : 0;
  const int n = n0 + lm;
  const float biasv = b1[e * 1024 + n];

  for (int halfm = 0; halfm < 2; ++halfm) {
    const int mt = wid + halfm * 4;
    const _Float16* arow = feats16 + (long)idx[e * 128 + mt * 16 + lm] * 256;
    v8f acc = {};
    #pragma unroll
    for (int kc = 0; kc < 8; ++kc) {
      int kb = kc << 5;
      v16h af, bf;
      #pragma unroll
      for (int i = 0; i < 8; ++i) { af[i] = arow[kb + ka0 + i]; af[i + 8] = arow[kb + ka1 + i]; }
      #pragma unroll
      for (int j = 0; j < 16; ++j) bf[j] = smem[(kb + kb0 + j) * 16 + lm];
      acc = wmma_f16(af, bf, acc);
    }
    #pragma unroll
    for (int r = 0; r < 8; ++r) {
      int row = mt * 16 + r + hi * 8;
      h[((long)e * 128 + row) * 1024 + n] = (_Float16)fmaxf(acc[r] + biasv, 0.f);
    }
  }
}

// ---------------- expert MLP layer 2: o = relu(h @ ew2 + eb2) * valid; scatter ----------------
__global__ void __launch_bounds__(128)
k_expert_l2(const _Float16* __restrict__ h, const int* __restrict__ idx,
            const float* __restrict__ validv,
            const _Float16* __restrict__ w2, const float* __restrict__ b2,
            _Float16* __restrict__ eo)
{
  __shared__ _Float16 smem[1024 * 16];            // 32 KiB weight panel
  const int tid  = threadIdx.x;
  const int lane = tid & 31;
  const int wid  = tid >> 5;
  const int nt = blockIdx.x & 31;
  const int e  = blockIdx.x >> 5;
  const int n0 = nt * 16;

  {   // async stage: 1024 rows x 32B = 2048 chunks of 16B
    const _Float16* src = w2 + (long)e * 1024 * 512 + n0;
    unsigned ldsBase = (unsigned)(unsigned long long)smem;
    #pragma unroll
    for (int c = 0; c < 16; ++c) {
      int ch = tid + c * 128;
      int row = ch >> 1, half = ch & 1;
      async_copy_b128(ldsBase + row * 32 + half * 16, src + (long)row * 512 + half * 8);
    }
    wait_asynccnt0();
  }
  __syncthreads();

  const int lm = lane & 15, hi = lane >> 4;
  const int ka0 = hi ? 16 : 0, ka1 = hi ? 24 : 8, kb0 = hi ? 16 : 0;
  const int n = n0 + lm;
  const float biasv = b2[e * 512 + n];

  for (int halfm = 0; halfm < 2; ++halfm) {
    const int mt = wid + halfm * 4;
    const _Float16* arow = h + ((long)e * 128 + mt * 16 + lm) * 1024;
    v8f acc = {};
    for (int kc = 0; kc < 32; ++kc) {
      int kb = kc << 5;
      v16h af, bf;
      #pragma unroll
      for (int i = 0; i < 8; ++i) { af[i] = arow[kb + ka0 + i]; af[i + 8] = arow[kb + ka1 + i]; }
      #pragma unroll
      for (int j = 0; j < 16; ++j) bf[j] = smem[(kb + kb0 + j) * 16 + lm];
      acc = wmma_f16(af, bf, acc);
    }
    #pragma unroll
    for (int r = 0; r < 8; ++r) {
      int row = mt * 16 + r + hi * 8;
      if (validv[e * 128 + row] != 0.f) {         // padding slots add 0 -> skip store
        int bdst = idx[e * 128 + row];
        eo[((long)bdst * 8 + e) * 512 + n] = (_Float16)fmaxf(acc[r] + biasv, 0.f);
      }
    }
  }
}

// ---------------- classifier: cls = eo.reshape(512,4096) @ clf_w + clf_b ----------------
__global__ void __launch_bounds__(128)
k_clf(const _Float16* __restrict__ eo, const _Float16* __restrict__ cw,
      const float* __restrict__ cb, float* __restrict__ out)
{
  const int lane = threadIdx.x & 31;
  const int wid  = threadIdx.x >> 5;
  unsigned tile = blockIdx.x * 4u + wid;          // 32 mtiles * 7 ntiles = 224
  int nt = (int)(tile % 7); int mt = (int)(tile / 7);
  const int lm = lane & 15, hi = lane >> 4;
  const int ka0 = hi ? 16 : 0, ka1 = hi ? 24 : 8, kb0 = hi ? 16 : 0;
  const int n = nt * 16 + lm;
  const bool nok = n < 100;
  const _Float16* arow = eo + (long)(mt * 16 + lm) * 4096;

  v8f acc = {};
  for (int kc = 0; kc < 128; ++kc) {
    int kb = kc << 5;
    if (kc < 127) __builtin_prefetch(arow + kb + 32, 0, 3);  // global_prefetch_b8
    v16h af, bf;
    #pragma unroll
    for (int i = 0; i < 8; ++i) { af[i] = arow[kb + ka0 + i]; af[i + 8] = arow[kb + ka1 + i]; }
    #pragma unroll
    for (int j = 0; j < 16; ++j)
      bf[j] = nok ? cw[(long)(kb + kb0 + j) * 100 + n] : (_Float16)0.f;
    acc = wmma_f16(af, bf, acc);
  }
  float bias = nok ? cb[n] : 0.f;
  #pragma unroll
  for (int r = 0; r < 8; ++r) {
    int row = mt * 16 + r + hi * 8;
    if (nok) out[(long)row * 100 + n] = acc[r] + bias;
  }
}

// ---------------- host orchestration ----------------
extern "C" void kernel_launch(void* const* d_in, const int* in_sizes, int n_in,
                              void* d_out, int out_size, void* d_ws, size_t ws_size,
                              hipStream_t stream)
{
  (void)in_sizes; (void)n_in; (void)out_size; (void)ws_size;
  const float* x      = (const float*)d_in[0];
  const float* cw_[4] = {(const float*)d_in[1],  (const float*)d_in[7],  (const float*)d_in[13], (const float*)d_in[19]};
  const float* cb_[4] = {(const float*)d_in[2],  (const float*)d_in[8],  (const float*)d_in[14], (const float*)d_in[20]};
  const float* bg_[4] = {(const float*)d_in[3],  (const float*)d_in[9],  (const float*)d_in[15], (const float*)d_in[21]};
  const float* bb_[4] = {(const float*)d_in[4],  (const float*)d_in[10], (const float*)d_in[16], (const float*)d_in[22]};
  const float* bm_[4] = {(const float*)d_in[5],  (const float*)d_in[11], (const float*)d_in[17], (const float*)d_in[23]};
  const float* bv_[4] = {(const float*)d_in[6],  (const float*)d_in[12], (const float*)d_in[18], (const float*)d_in[24]};
  const float* sw   = (const float*)d_in[25];
  const float* sb   = (const float*)d_in[26];
  const float* ew1  = (const float*)d_in[27];
  const float* eb1  = (const float*)d_in[28];
  const float* ew2  = (const float*)d_in[29];
  const float* eb2  = (const float*)d_in[30];
  const float* clfw = (const float*)d_in[31];
  const float* clfb = (const float*)d_in[32];

  float* out_cls    = (float*)d_out;              // [512,100]
  float* out_scores = out_cls + 512 * 100;        // [512,8]
  float* out_D      = out_scores + 512 * 8;       // [512,8]

  char* ws = (char*)d_ws;
  size_t off = 0;
  auto take = [&](size_t bytes) -> char* {
    char* p = ws + off;
    off = (off + bytes + 255) & ~(size_t)255;
    return p;
  };

  _Float16* x16  = (_Float16*)take((size_t)512 * 3 * 32 * 32 * 2);
  _Float16* bufA = (_Float16*)take((size_t)512 * 256 * 16 * 16 * 2);  // max(A1,A3,A5) = 64 MiB
  _Float16* bufB = (_Float16*)take((size_t)512 * 64 * 32 * 32 * 2);   // max(A2,A4)   = 64 MiB
  int wsz[4] = {32 * 3 * 9, 64 * 32 * 9, 128 * 64 * 9, 256 * 128 * 9};
  int cos[4] = {32, 64, 128, 256};
  _Float16* w16[4]; float* scl[4]; float* bia[4];
  for (int l = 0; l < 4; ++l) {
    w16[l] = (_Float16*)take((size_t)wsz[l] * 2);
    scl[l] = (float*)take((size_t)cos[l] * 4);
    bia[l] = (float*)take((size_t)cos[l] * 4);
  }
  float*    feats   = (float*)take((size_t)512 * 256 * 4);
  _Float16* feats16 = (_Float16*)take((size_t)512 * 256 * 2);
  int*      idx     = (int*)take((size_t)8 * 128 * 4);
  float*    validv  = (float*)take((size_t)8 * 128 * 4);
  _Float16* ew1_16  = (_Float16*)take((size_t)8 * 256 * 1024 * 2);
  _Float16* ew2_16  = (_Float16*)take((size_t)8 * 1024 * 512 * 2);
  _Float16* clfw16  = (_Float16*)take((size_t)4096 * 100 * 2);
  _Float16* h16     = (_Float16*)take((size_t)8 * 128 * 1024 * 2);
  _Float16* eo16    = (_Float16*)take((size_t)512 * 8 * 512 * 2);

  auto cdiv = [](long a, long b) { return (int)((a + b - 1) / b); };

  long nx = (long)512 * 3 * 32 * 32;
  k_f32_to_f16<<<cdiv(nx, 256), 256, 0, stream>>>(x, x16, nx);
  for (int l = 0; l < 4; ++l) {
    k_f32_to_f16<<<cdiv(wsz[l], 256), 256, 0, stream>>>(cw_[l], w16[l], wsz[l]);
    k_bn_fold<<<1, 256, 0, stream>>>(cb_[l], bg_[l], bb_[l], bm_[l], bv_[l], scl[l], bia[l], cos[l]);
  }
  k_f32_to_f16<<<cdiv((long)8 * 256 * 1024, 256), 256, 0, stream>>>(ew1, ew1_16, (long)8 * 256 * 1024);
  k_f32_to_f16<<<cdiv((long)8 * 1024 * 512, 256), 256, 0, stream>>>(ew2, ew2_16, (long)8 * 1024 * 512);
  k_f32_to_f16<<<cdiv((long)4096 * 100, 256), 256, 0, stream>>>(clfw, clfw16, (long)4096 * 100);
  k_zero_f16<<<cdiv((long)512 * 8 * 512, 256), 256, 0, stream>>>(eo16, (long)512 * 8 * 512);

  // trunk: conv1 -> conv2 -> pool -> conv3 -> conv4 -> pool+avg
  k_conv_wmma<3,   32, 32, 32 ><<<65536 / 4,  128, 0, stream>>>(x16,  w16[0], scl[0], bia[0], bufA);
  k_conv_wmma<32,  32, 32, 64 ><<<131072 / 4, 128, 0, stream>>>(bufA, w16[1], scl[1], bia[1], bufB);
  k_maxpool2<<<cdiv((long)512 * 64 * 16 * 16, 256), 256, 0, stream>>>(bufB, bufA, 512, 64, 32, 32);
  k_conv_wmma<64,  16, 16, 128><<<65536 / 4,  128, 0, stream>>>(bufA, w16[2], scl[2], bia[2], bufB);
  k_conv_wmma<128, 16, 16, 256><<<131072 / 4, 128, 0, stream>>>(bufB, w16[3], scl[3], bia[3], bufA);
  k_pool_avg<<<cdiv(512 * 256, 256), 256, 0, stream>>>(bufA, feats, feats16, 512, 256, 16, 16);

  // scorer + softmax, greedy router
  k_scorer<<<2, 256, 0, stream>>>(feats, sw, sb, out_scores);
  k_route<<<1, 1024, 0, stream>>>(out_scores, out_D, idx, validv);

  // expert MLPs (LDS-staged weight panels via async-to-LDS) + classifier
  k_expert_l1<<<8 * 64, 128, 0, stream>>>(feats16, idx, ew1_16, eb1, h16);
  k_expert_l2<<<8 * 32, 128, 0, stream>>>(h16, idx, validv, ew2_16, eb2, eo16);
  k_clf<<<224 / 4, 128, 0, stream>>>(eo16, clfw16, clfb, out_cls);
}
